// GlobalAttention_50603304681504
// MI455X (gfx1250) — matrix operand order repaired and checked by reading, not compiled, not run on hardware
//
#include <hip/hip_runtime.h>
#include <hip/hip_bf16.h>

// MI455X / gfx1250: GroupNorm + 8-head global attention + proj.
// All GEMMs on v_wmma_f32_16x16x32_f16; flash-style streaming softmax with
// 64-key steps; global_prefetch_b8 for next key block.

typedef _Float16 half8 __attribute__((ext_vector_type(8)));
typedef _Float16 v16h  __attribute__((ext_vector_type(16)));
typedef float    v8f   __attribute__((ext_vector_type(8)));

#define C_DIM 256
#define N_SP  4096   // 64*64 spatial positions
#define HD_DIM 32    // head dim

static __device__ __forceinline__ v8f wmma16(v16h a, v16h b, v8f c) {
  // D = A(16x32 f16) * B(32x16 f16) + C(16x16 f32)
  return __builtin_amdgcn_wmma_f32_16x16x32_f16(false, a, false, b, (short)0, c,
                                                false, false);
}

// A operand (16x32, f16), memory row-major [row][k] with k contiguous.
// Lane group g = lane/16 holds K = {g*8..g*8+7} and {16+g*8..16+g*8+7}.
static __device__ __forceinline__ v16h load_A16(const _Float16* __restrict__ base,
                                                int stride, int row, int kbase,
                                                int lane) {
  const int g = (lane >> 4) & 1;
  const _Float16* p = base + (size_t)row * stride + kbase + g * 8;
  half8 lo = *(const half8*)(p);
  half8 hi = *(const half8*)(p + 16);
  v16h r;
#pragma unroll
  for (int i = 0; i < 8; ++i) { r[i] = lo[i]; r[i + 8] = hi[i]; }
  return r;
}

// B operand (32x16, f16), memory row-major [n][k] with k contiguous.
// Lane group g = lane/16 holds K = {g*8..g*8+7} and {16+g*8..16+g*8+7}.
static __device__ __forceinline__ v16h load_B16(const _Float16* __restrict__ base,
                                                int stride, int row, int kbase,
                                                int lane) {
  const int g = (lane >> 4) & 1;
  const _Float16* p = base + (size_t)row * stride + kbase + g * 8;
  half8 lo = *(const half8*)(p);
  half8 hi = *(const half8*)(p + 16);
  v16h r;
#pragma unroll
  for (int i = 0; i < 8; ++i) { r[i] = lo[i]; r[i + 8] = hi[i]; }
  return r;
}

// ---------------------------------------------------------------- fp32->fp16
__global__ __launch_bounds__(256) void cvt_f32_f16(const float* __restrict__ a,
                                                   _Float16* __restrict__ b, int n) {
  int i = blockIdx.x * 256 + threadIdx.x;
  if (i < n) b[i] = (_Float16)a[i];
}

// ---------------------------------------------------------------- GroupNorm
// One block per group (8 groups x 32 channels x 4096). Writes xnT[p][c] (f16),
// channel-contiguous so it serves as a WMMA B operand directly.
__global__ __launch_bounds__(256) void groupnorm_kernel(
    const float* __restrict__ x, const float* __restrict__ gns,
    const float* __restrict__ gnb, _Float16* __restrict__ xnT) {
  const int g = blockIdx.x;
  const int tid = threadIdx.x;
  const size_t base = (size_t)g * 32 * N_SP;
  float s = 0.f, s2 = 0.f;
  for (int i = tid; i < 32 * N_SP; i += 256) {
    float v = x[base + i];
    s += v; s2 += v * v;
  }
#pragma unroll
  for (int off = 16; off >= 1; off >>= 1) {
    s  += __shfl_xor(s,  off, 32);
    s2 += __shfl_xor(s2, off, 32);
  }
  __shared__ float ss[8], ss2[8];
  const int wid = tid >> 5, lane = tid & 31;
  if (lane == 0) { ss[wid] = s; ss2[wid] = s2; }
  __syncthreads();
  if (tid == 0) {
    float a = 0.f, b = 0.f;
#pragma unroll
    for (int i = 0; i < 8; ++i) { a += ss[i]; b += ss2[i]; }
    ss[0] = a; ss2[0] = b;
  }
  __syncthreads();
  const float inv_n = 1.0f / (32.0f * N_SP);
  const float mean = ss[0] * inv_n;
  const float var  = ss2[0] * inv_n - mean * mean;
  const float rstd = rsqrtf(var + 1e-5f);
  for (int i = tid; i < 32 * N_SP; i += 256) {
    const int cl = i >> 12;         // / 4096
    const int p  = i & (N_SP - 1);
    const int c  = g * 32 + cl;
    float v = (x[base + i] - mean) * rstd * gns[c] + gnb[c];
    xnT[(size_t)p * C_DIM + c] = (_Float16)v;
  }
}

// ---------------------------------------------------------------- QKV GEMMs
// M=256 (o) x N=4096 (p) x K=256 (c), three matrices. Each wave: 16(o)x64(p).
// q,k stored transposed [p][c] with scale folded into q; v stored [c][p].
__global__ __launch_bounds__(128) void qkv_gemm_kernel(
    const _Float16* __restrict__ wq16, const _Float16* __restrict__ wk16,
    const _Float16* __restrict__ wv16, const float* __restrict__ bq,
    const float* __restrict__ bk, const float* __restrict__ bv,
    const _Float16* __restrict__ xnT, _Float16* __restrict__ qT,
    _Float16* __restrict__ kT, _Float16* __restrict__ vbuf) {
  const int lane = threadIdx.x & 31;
  const int w = (blockIdx.x * 128 + threadIdx.x) >> 5;  // 0..3071
  const int mat = w >> 10;                              // 0=q 1=k 2=v
  const int rem = w & 1023;
  const int ot = (rem >> 6) << 4;                       // o tile base
  const int pb = (rem & 63) << 6;                       // p base (64-wide)
  const _Float16* W = (mat == 0) ? wq16 : (mat == 1) ? wk16 : wv16;
  const float* bias = (mat == 0) ? bq : (mat == 1) ? bk : bv;

  v8f acc0 = {0,0,0,0,0,0,0,0}, acc1 = acc0, acc2 = acc0, acc3 = acc0;
  for (int kk = 0; kk < C_DIM; kk += 32) {
    v16h a  = load_A16(W,   C_DIM, ot + (lane & 15), kk, lane);
    v16h b0 = load_B16(xnT, C_DIM, pb +  0 + (lane & 15), kk, lane);
    v16h b1 = load_B16(xnT, C_DIM, pb + 16 + (lane & 15), kk, lane);
    v16h b2 = load_B16(xnT, C_DIM, pb + 32 + (lane & 15), kk, lane);
    v16h b3 = load_B16(xnT, C_DIM, pb + 48 + (lane & 15), kk, lane);
    acc0 = wmma16(a, b0, acc0);
    acc1 = wmma16(a, b1, acc1);
    acc2 = wmma16(a, b2, acc2);
    acc3 = wmma16(a, b3, acc3);
  }
  const int og = ot + ((lane >> 4) << 3);  // this lane's 8 consecutive o rows
  const float scale = 0.17677669529663687f;  // HD^-0.5
  v8f accs[4] = {acc0, acc1, acc2, acc3};
#pragma unroll
  for (int j = 0; j < 4; ++j) {
    const int p = pb + j * 16 + (lane & 15);
    if (mat == 2) {
#pragma unroll
      for (int r = 0; r < 8; ++r)
        vbuf[(size_t)(og + r) * N_SP + p] = (_Float16)(accs[j][r] + bias[og + r]);
    } else {
      half8 h;
#pragma unroll
      for (int r = 0; r < 8; ++r) {
        float t = accs[j][r] + bias[og + r];
        if (mat == 0) t *= scale;
        h[r] = (_Float16)t;
      }
      _Float16* dst = ((mat == 0) ? qT : kT) + (size_t)p * C_DIM + og;
      *(half8*)dst = h;                    // 16B contiguous store
    }
  }
}

// ---------------------------------------------------------------- attention
// One wave per (head, 16-query tile): flash-style over 4096 keys in steps of
// 64. Per step: 4 S-WMMAs (K=head_dim=32), ONE online-softmax update per row
// (halves shfl-reduction cost vs 32-key steps), P relayout D->A through a
// private LDS strip, 4 PV-WMMAs. Next block's K/V lines are prefetched.
__global__ __launch_bounds__(128) void attention_kernel(
    const _Float16* __restrict__ qT, const _Float16* __restrict__ kT,
    const _Float16* __restrict__ vbuf, _Float16* __restrict__ attnT) {
  const int lane = threadIdx.x & 31;
  const int wid = threadIdx.x >> 5;
  const int w = blockIdx.x * 4 + wid;        // 0..2047
  const int h = w >> 8;                      // head
  const int qbase = (w & 255) << 4;          // query tile base
  const int lg = (lane >> 4) & 1;            // lane group

  __shared__ _Float16 pbuf_s[4][16 * 64];
  _Float16* pbuf = pbuf_s[wid];

  const v8f zero = {0,0,0,0,0,0,0,0};
  // Q^T as A operand: rows = queries, K = head dim (contiguous in qT[p][c])
  const v16h aq = load_A16(qT, C_DIM, qbase + (lane & 15), h * HD_DIM, lane);

  v8f o1 = zero, o2 = zero;
  float mrow[8], lrow[8];
#pragma unroll
  for (int r = 0; r < 8; ++r) { mrow[r] = -1e30f; lrow[r] = 0.f; }

  for (int m0 = 0; m0 < N_SP; m0 += 64) {
    if (m0 + 64 < N_SP) {
      // Next key block: K rows m0+64..m0+127 (lane-spread), V lines (128B
      // covers 64 keys). Emits global_prefetch_b8; no counter cost.
      __builtin_prefetch(kT + (size_t)(m0 + 64 + lane) * C_DIM + h * HD_DIM, 0, 1);
      __builtin_prefetch(kT + (size_t)(m0 + 96 + lane) * C_DIM + h * HD_DIM, 0, 1);
      __builtin_prefetch(vbuf + (size_t)(h * HD_DIM + lane) * N_SP + m0 + 64, 0, 1);
    }
    // Scores for 64 keys: 4 WMMAs, K-dim = head dim (contiguous in kT[m][c])
    v8f s[4];
#pragma unroll
    for (int jj = 0; jj < 4; ++jj) {
      v16h bk = load_B16(kT, C_DIM, m0 + jj * 16 + (lane & 15), h * HD_DIM, lane);
      s[jj] = wmma16(aq, bk, zero);
    }

#pragma unroll
    for (int r = 0; r < 8; ++r) {
      // row = r + 8*lg is uniform across this lane's 16-lane column group
      float mx = fmaxf(fmaxf(s[0][r], s[1][r]), fmaxf(s[2][r], s[3][r]));
#pragma unroll
      for (int off = 8; off >= 1; off >>= 1) mx = fmaxf(mx, __shfl_xor(mx, off, 32));
      const float mnew  = fmaxf(mrow[r], mx);
      const float alpha = __expf(mrow[r] - mnew);
      float pr[4];
#pragma unroll
      for (int jj = 0; jj < 4; ++jj) pr[jj] = __expf(s[jj][r] - mnew);
      float rs = (pr[0] + pr[1]) + (pr[2] + pr[3]);
#pragma unroll
      for (int off = 8; off >= 1; off >>= 1) rs += __shfl_xor(rs, off, 32);
      lrow[r] = lrow[r] * alpha + rs;
      mrow[r] = mnew;
      o1[r] *= alpha; o2[r] *= alpha;
      const int n = r + (lg << 3);           // D-layout row for this lane
#pragma unroll
      for (int jj = 0; jj < 4; ++jj)
        pbuf[n * 64 + jj * 16 + (lane & 15)] = (_Float16)pr[jj];
    }
    asm volatile("s_wait_dscnt 0" ::: "memory");  // P stores -> A reads (same wave)

    // P as two A operands (16 queries x 32 keys each) from LDS
    v16h ap0, ap1;
    {
      const _Float16* pp = pbuf + (lane & 15) * 64 + lg * 8;
      half8 lo0 = *(const half8*)(pp);
      half8 hi0 = *(const half8*)(pp + 16);
      half8 lo1 = *(const half8*)(pp + 32);
      half8 hi1 = *(const half8*)(pp + 48);
#pragma unroll
      for (int i = 0; i < 8; ++i) {
        ap0[i] = lo0[i]; ap0[i + 8] = hi0[i];
        ap1[i] = lo1[i]; ap1[i + 8] = hi1[i];
      }
    }
    // V^T as B operands: rows = channels (d), K-dim = keys (contig in v[c][p])
    v16h bv00 = load_B16(vbuf, N_SP, h * HD_DIM +      (lane & 15), m0,      lane);
    v16h bv01 = load_B16(vbuf, N_SP, h * HD_DIM + 16 + (lane & 15), m0,      lane);
    v16h bv10 = load_B16(vbuf, N_SP, h * HD_DIM +      (lane & 15), m0 + 32, lane);
    v16h bv11 = load_B16(vbuf, N_SP, h * HD_DIM + 16 + (lane & 15), m0 + 32, lane);
    o1 = wmma16(ap0, bv00, o1);   // keys 0..31  -> d 0..15
    o2 = wmma16(ap0, bv01, o2);   // keys 0..31  -> d 16..31
    o1 = wmma16(ap1, bv10, o1);   // keys 32..63 -> d 0..15
    o2 = wmma16(ap1, bv11, o2);   // keys 32..63 -> d 16..31
  }

#pragma unroll
  for (int r = 0; r < 8; ++r) {
    const int n = qbase + r + (lg << 3);
    const float inv = 1.0f / lrow[r];
    const int d = lane & 15;
    attnT[(size_t)n * C_DIM + h * HD_DIM + d]      = (_Float16)(o1[r] * inv);
    attnT[(size_t)n * C_DIM + h * HD_DIM + 16 + d] = (_Float16)(o2[r] * inv);
  }
}

// ---------------------------------------------------------------- out proj
// y = x + gamma * (Wo @ attn + bo). M=256 x N=4096 x K=256, 16x64 per wave.
__global__ __launch_bounds__(128) void out_proj_kernel(
    const _Float16* __restrict__ wo16, const float* __restrict__ bo,
    const _Float16* __restrict__ attnT, const float* __restrict__ x,
    const float* __restrict__ gamma, float* __restrict__ out) {
  const int lane = threadIdx.x & 31;
  const int w = (blockIdx.x * 128 + threadIdx.x) >> 5;  // 0..1023
  const int ot = (w >> 6) << 4;
  const int pb = (w & 63) << 6;

  v8f acc0 = {0,0,0,0,0,0,0,0}, acc1 = acc0, acc2 = acc0, acc3 = acc0;
  for (int kk = 0; kk < C_DIM; kk += 32) {
    v16h a  = load_A16(wo16,  C_DIM, ot + (lane & 15), kk, lane);
    v16h b0 = load_B16(attnT, C_DIM, pb +  0 + (lane & 15), kk, lane);
    v16h b1 = load_B16(attnT, C_DIM, pb + 16 + (lane & 15), kk, lane);
    v16h b2 = load_B16(attnT, C_DIM, pb + 32 + (lane & 15), kk, lane);
    v16h b3 = load_B16(attnT, C_DIM, pb + 48 + (lane & 15), kk, lane);
    acc0 = wmma16(a, b0, acc0);
    acc1 = wmma16(a, b1, acc1);
    acc2 = wmma16(a, b2, acc2);
    acc3 = wmma16(a, b3, acc3);
  }
  const int og = ot + ((lane >> 4) << 3);
  const float gm = gamma[0];
  v8f accs[4] = {acc0, acc1, acc2, acc3};
#pragma unroll
  for (int j = 0; j < 4; ++j) {
    const int p = pb + j * 16 + (lane & 15);
#pragma unroll
    for (int r = 0; r < 8; ++r) {
      const size_t idx = (size_t)(og + r) * N_SP + p;
      out[idx] = x[idx] + gm * (accs[j][r] + bo[og + r]);
    }
  }
}

// ---------------------------------------------------------------- launch
extern "C" void kernel_launch(void* const* d_in, const int* in_sizes, int n_in,
                              void* d_out, int out_size, void* d_ws, size_t ws_size,
                              hipStream_t stream) {
  const float* x     = (const float*)d_in[0];
  const float* wq    = (const float*)d_in[1];
  const float* bq    = (const float*)d_in[2];
  const float* wk    = (const float*)d_in[3];
  const float* bk    = (const float*)d_in[4];
  const float* wv    = (const float*)d_in[5];
  const float* bv    = (const float*)d_in[6];
  const float* wo    = (const float*)d_in[7];
  const float* bo    = (const float*)d_in[8];
  const float* gns   = (const float*)d_in[9];
  const float* gnb   = (const float*)d_in[10];
  const float* gamma = (const float*)d_in[11];
  float* out = (float*)d_out;

  _Float16* ws    = (_Float16*)d_ws;       // ~10.5 MB used
  _Float16* wq16  = ws;
  _Float16* wk16  = wq16 + 65536;
  _Float16* wv16  = wk16 + 65536;
  _Float16* wo16  = wv16 + 65536;
  _Float16* xnT   = wo16 + 65536;          // [4096][256]
  _Float16* qT    = xnT  + 1048576;        // [4096][256] (scaled, +bias)
  _Float16* kT    = qT   + 1048576;        // [4096][256]
  _Float16* vbuf  = kT   + 1048576;        // [256][4096]
  _Float16* attnT = vbuf + 1048576;        // [4096][256]

  cvt_f32_f16<<<256, 256, 0, stream>>>(wq, wq16, 65536);
  cvt_f32_f16<<<256, 256, 0, stream>>>(wk, wk16, 65536);
  cvt_f32_f16<<<256, 256, 0, stream>>>(wv, wv16, 65536);
  cvt_f32_f16<<<256, 256, 0, stream>>>(wo, wo16, 65536);
  groupnorm_kernel<<<8, 256, 0, stream>>>(x, gns, gnb, xnT);
  qkv_gemm_kernel<<<768, 128, 0, stream>>>(wq16, wk16, wv16, bq, bk, bv,
                                           xnT, qT, kT, vbuf);
  attention_kernel<<<512, 128, 0, stream>>>(qT, kT, vbuf, attnT);
  out_proj_kernel<<<256, 128, 0, stream>>>(wo16, bo, attnT, x, gamma, out);
}
